// SimpleLSTM_58523224375384
// MI455X (gfx1250) — matrix operand order repaired and checked by reading, not compiled, not run on hardware
//
#include <hip/hip_runtime.h>

// ----------------------------------------------------------------------------
// SimpleLSTM on MI455X (gfx1250, wave32, WMMA).
//
// Phase 1: xproj = x@Wx + b   (big parallel bf16 WMMA GEMM, 275 GFLOP)
// Phase 2: 512 stream-ordered step kernels  z = xproj_t + h@Wh  (bf16 WMMA,
//          Wh bf16 = 8MB -> L2-resident across the whole recurrence)
// Phase 3: y = h_seq@Wy + b   (big parallel bf16 WMMA GEMM)
// All accumulation / gates / cell state in fp32; operands bf16.
// Big GEMMs use double-buffered LDS panels fed by GLOBAL_LOAD_ASYNC_TO_LDS_B128
// (ASYNCcnt pipeline, one barrier per K-chunk) when the builtin is available.
// ----------------------------------------------------------------------------

#define BATCH 64
#define TSTEPS 512
#define DDIM 1024
#define HDIM 1024
#define GDIM 4096   // 4*HDIM
#define ODIM 1024

#if defined(__has_builtin)
# if __has_builtin(__builtin_amdgcn_global_load_async_to_lds_b128)
#  define USE_ASYNC_LDS 1
# endif
#endif

typedef __attribute__((ext_vector_type(16))) __bf16 bf16x16;
typedef __attribute__((ext_vector_type(8)))  float  v8f;
typedef int v4i __attribute__((vector_size(16)));

// async-copy pointer casts: AS1 (__device__) source, AS3 (__shared__) dest,
// pointee type = 16-byte int vector (exact builtin signature per clang).
#define GLOBAL_V4I(p) ((__attribute__((address_space(1))) v4i*)(p))
#define LDS_V4I(p)    ((__attribute__((address_space(3))) v4i*)(p))

union Frag {
  bf16x16 v;
  unsigned int u[8];
};

static __device__ __forceinline__ unsigned short f2bf(float f) {
  union { float f; unsigned int u; } x; x.f = f;
  unsigned int u = x.u;
  unsigned int r = u + 0x7FFFu + ((u >> 16) & 1u);   // round-to-nearest-even
  return (unsigned short)(r >> 16);
}

static __device__ __forceinline__ float sigmoidf_(float z) {
  return 1.0f / (1.0f + __expf(-z));
}

static __device__ __forceinline__ void async_wait0() {
#if defined(__has_builtin) && __has_builtin(__builtin_amdgcn_s_wait_asynccnt)
  __builtin_amdgcn_s_wait_asynccnt(0);
#else
  asm volatile("s_wait_asynccnt 0x0" ::: "memory");
#endif
}

// ---------------------------------------------------------------------------
// Shared WMMA panel core.  Computes a 64x64 C-tile of  A[m0..m0+64) x
// Bt[n0..n0+64)^T  with K=1024, lda=ldb=1024 (bf16, k-contiguous both sides).
// 8 waves: wave w owns subtile row (w&3), subtile cols {2*(w>>2), 2*(w>>2)+1}.
// LDS panels are [64 rows][32 bf16] (k-contiguous), double buffered.
// ---------------------------------------------------------------------------
static __device__ __forceinline__ void wmma_compute_panel(
    const unsigned int* __restrict__ As, const unsigned int* __restrict__ Bs,
    int mrow, int nc0, int lo, int hi, v8f& acc0, v8f& acc1)
{
  Frag fa, fb0, fb1;
  const unsigned int* ar  = &As[(mrow + lo) * 16];
  const unsigned int* br0 = &Bs[(nc0 + lo) * 16];
  const unsigned int* br1 = &Bs[(nc0 + 16 + lo) * 16];
#pragma unroll
  for (int v = 0; v < 4; ++v) {          // A 16x32 bf16, k = 16*hi + s (matches B)
    fa.u[v]     = ar[v + 8 * hi];
    fa.u[v + 4] = ar[4 + v + 8 * hi];
  }
#pragma unroll
  for (int v = 0; v < 8; ++v) {          // B 32x16 bf16 layout
    fb0.u[v] = br0[v + 8 * hi];
    fb1.u[v] = br1[v + 8 * hi];
  }
  acc0 = __builtin_amdgcn_wmma_f32_16x16x32_bf16(false, fa.v, false, fb0.v,
                                                 (short)0, acc0, false, false);
  acc1 = __builtin_amdgcn_wmma_f32_16x16x32_bf16(false, fa.v, false, fb1.v,
                                                 (short)0, acc1, false, false);
}

static __device__ __forceinline__ void wmma_panel_gemm(
    const unsigned short* __restrict__ A,   // rows m0..m0+63, stride 1024
    const unsigned short* __restrict__ Bt,  // rows n0..n0+63, stride 1024
    int m0, int n0,
    unsigned int* As, unsigned int* Bs,     // LDS: 2 x 1024 u32 each
    v8f& acc0, v8f& acc1)
{
  const int tid = threadIdx.x;
  const int w = tid >> 5, lane = tid & 31;
  const int lo = lane & 15, hi = lane >> 4;
  const int mrow = (w & 3) * 16;
  const int nc0  = (w >> 2) * 32;
  const int srow = tid >> 2, sq = tid & 3;        // staging: 16B per thread
  const size_t aoff = (size_t)(m0 + srow) * 1024 + sq * 8;
  const size_t boff = (size_t)(n0 + srow) * 1024 + sq * 8;

#ifdef USE_ASYNC_LDS
  // ---- async double-buffered pipeline: 1 barrier per K-chunk ----
  __builtin_amdgcn_global_load_async_to_lds_b128(
      GLOBAL_V4I(A + aoff), LDS_V4I(As + tid * 4), 0, 0);
  __builtin_amdgcn_global_load_async_to_lds_b128(
      GLOBAL_V4I(Bt + boff), LDS_V4I(Bs + tid * 4), 0, 0);
  int cur = 0;
  for (int i = 0; i < 32; ++i) {
    async_wait0();                 // tile i landed in buffer `cur`
    __syncthreads();               // everyone done reading buffer cur^1
    if (i + 1 < 32) {              // prefetch tile i+1 into buffer cur^1
      const int kb = (i + 1) * 32;
      __builtin_amdgcn_global_load_async_to_lds_b128(
          GLOBAL_V4I(A + aoff + kb),
          LDS_V4I(As + ((cur ^ 1) << 10) + tid * 4), 0, 0);
      __builtin_amdgcn_global_load_async_to_lds_b128(
          GLOBAL_V4I(Bt + boff + kb),
          LDS_V4I(Bs + ((cur ^ 1) << 10) + tid * 4), 0, 0);
    }
    wmma_compute_panel(As + (cur << 10), Bs + (cur << 10), mrow, nc0, lo, hi, acc0, acc1);
    cur ^= 1;
  }
#else
  // ---- fallback: register-staged, b128 transfers, 2 barriers per chunk ----
  for (int kb = 0; kb < 1024; kb += 32) {
    const uint4 av = *(const uint4*)(A + aoff + kb);
    const uint4 bv = *(const uint4*)(Bt + boff + kb);
    ((uint4*)As)[tid] = av;
    ((uint4*)Bs)[tid] = bv;
    __syncthreads();
    wmma_compute_panel(As, Bs, mrow, nc0, lo, hi, acc0, acc1);
    __syncthreads();
  }
#endif
}

// ---------------------------------------------------------------------------
// Weight packing: transposed bf16 weights so WMMA A/B fragments are
// k-contiguous (one u32 per bf16 pair).
// ---------------------------------------------------------------------------
__global__ __launch_bounds__(256) void lstm_pack_weights(
    const float* __restrict__ Wf, const float* __restrict__ Wi,
    const float* __restrict__ Wc, const float* __restrict__ Wo,
    const float* __restrict__ bf, const float* __restrict__ bi,
    const float* __restrict__ bc, const float* __restrict__ bo,
    const float* __restrict__ Wy,
    unsigned short* __restrict__ WxT, unsigned short* __restrict__ WhT,
    unsigned short* __restrict__ WyT, float* __restrict__ ball)
{
  const int gcol = blockIdx.x;              // 0..4095
  const int gate = gcol >> 10;
  const int j    = gcol & 1023;
  const float* W = (gate == 0) ? Wf : (gate == 1) ? Wi : (gate == 2) ? Wc : Wo;
  for (int d = threadIdx.x; d < DDIM; d += 256) {
    WxT[(size_t)gcol * DDIM + d] = f2bf(W[(size_t)d * HDIM + j]);
    WhT[(size_t)gcol * HDIM + d] = f2bf(W[(size_t)(DDIM + d) * HDIM + j]);
  }
  if (threadIdx.x == 0) {
    const float* bb = (gate == 0) ? bf : (gate == 1) ? bi : (gate == 2) ? bc : bo;
    ball[gcol] = bb[j];
  }
  if (gcol < ODIM) {
    for (int h = threadIdx.x; h < HDIM; h += 256)
      WyT[(size_t)gcol * HDIM + h] = f2bf(Wy[(size_t)h * ODIM + gcol]);
  }
}

__global__ void lstm_convert_x(const float* __restrict__ x,
                               unsigned short* __restrict__ xbf, size_t n)
{
  size_t i = (size_t)blockIdx.x * blockDim.x + threadIdx.x;
  const size_t stride = (size_t)gridDim.x * blockDim.x;
  for (; i < n; i += stride) xbf[i] = f2bf(x[i]);
}

__global__ void lstm_init_state(float* __restrict__ h, float* __restrict__ c,
                                unsigned short* __restrict__ hbf0,
                                unsigned short* __restrict__ hbf1)
{
  const int i = blockIdx.x * blockDim.x + threadIdx.x;  // 65536 threads
  h[i] = 0.0f; c[i] = 0.0f; hbf0[i] = 0; hbf1[i] = 0;
}

// ---------------------------------------------------------------------------
// Phase 1: xproj[t*64+b][g] = sum_d x[b][t][d]*WxT[g][d] + ball[g]
// M=32768 (rows m=b*512+t), N=4096, K=1024.
// ---------------------------------------------------------------------------
__global__ __launch_bounds__(256) void lstm_gemm_xproj(
    const unsigned short* __restrict__ xbf,
    const unsigned short* __restrict__ WxT,
    const float* __restrict__ ball,
    float* __restrict__ xproj)
{
  __shared__ __align__(16) unsigned int As[2 * 64 * 16];
  __shared__ __align__(16) unsigned int Bs[2 * 64 * 16];
  const int m0 = blockIdx.x * 64, n0 = blockIdx.y * 64;
  const int tid = threadIdx.x;
  const int w = tid >> 5, lane = tid & 31;
  const int lo = lane & 15, hi = lane >> 4;
  const int mrow = (w & 3) * 16;
  const int nc0  = (w >> 2) * 32;

  v8f acc0 = {}, acc1 = {};
  wmma_panel_gemm(xbf, WxT, m0, n0, As, Bs, acc0, acc1);

  // epilogue: rows m=b*512+t  ->  xproj row r=t*64+b, add bias
#pragma unroll
  for (int i = 0; i < 8; ++i) {
    const int gm = m0 + mrow + 8 * hi + i;
    const int b = gm >> 9, t = gm & 511;
    const size_t r = (size_t)(t * 64 + b);
    const int col0 = n0 + nc0 + lo;
    const int col1 = col0 + 16;
    xproj[r * GDIM + col0] = acc0[i] + ball[col0];
    xproj[r * GDIM + col1] = acc1[i] + ball[col1];
  }
}

// ---------------------------------------------------------------------------
// Phase 2: one launch per timestep; grid (64,4) = (ntile, mtile).
// 4 waves = 4 gates, each one 16x16 WMMA z-tile (K=1024); gates combined
// through LDS; fp32 cell update; double-buffered bf16 h.
// ---------------------------------------------------------------------------
__global__ __launch_bounds__(128) void lstm_step(
    const unsigned short* __restrict__ hbf_in,
    const unsigned short* __restrict__ WhT,
    const float* __restrict__ xproj_t,       // xproj + t*64*4096
    float* __restrict__ c, float* __restrict__ h,
    unsigned short* __restrict__ hbf_out,
    unsigned short* __restrict__ hseq_t)     // hseq_bf + t*64*1024
{
  __shared__ float zs[4][16][16];
  const int ntile = blockIdx.x;              // 0..63
  const int mtile = blockIdx.y;              // 0..3
  const int tid = threadIdx.x;
  const int g = tid >> 5, lane = tid & 31;
  const int lo = lane & 15, hi = lane >> 4;

  const unsigned short* arow = hbf_in + (size_t)(mtile * 16 + lo) * HDIM;
  const unsigned short* brow = WhT + (size_t)(g * HDIM + ntile * 16 + lo) * HDIM;

  v8f acc = {};
  for (int kb = 0; kb < HDIM; kb += 32) {
    Frag fa, fb;
#pragma unroll
    for (int v = 0; v < 4; ++v) {
      fa.u[v]     = *(const unsigned int*)(arow + kb + 2 * v + 16 * hi);
      fa.u[v + 4] = *(const unsigned int*)(arow + kb + 8 + 2 * v + 16 * hi);
    }
#pragma unroll
    for (int v = 0; v < 8; ++v)
      fb.u[v] = *(const unsigned int*)(brow + kb + 2 * v + 16 * hi);
    acc = __builtin_amdgcn_wmma_f32_16x16x32_bf16(false, fa.v, false, fb.v,
                                                  (short)0, acc, false, false);
  }
#pragma unroll
  for (int i = 0; i < 8; ++i) {
    const int m = 8 * hi + i;
    const int row = mtile * 16 + m;                    // batch index
    const int gcol = g * HDIM + ntile * 16 + lo;
    zs[g][m][lo] = acc[i] + xproj_t[(size_t)row * GDIM + gcol];
  }
  __syncthreads();
#pragma unroll
  for (int e = 0; e < 2; ++e) {
    const int idx = tid + e * 128;                     // 256 state elements
    const int m = idx >> 4, n = idx & 15;
    const float ft = sigmoidf_(zs[0][m][n]);
    const float it = sigmoidf_(zs[1][m][n]);
    const float ct = tanhf(zs[2][m][n]);
    const float ot = sigmoidf_(zs[3][m][n]);
    const int row = mtile * 16 + m, col = ntile * 16 + n;
    const size_t s = (size_t)row * HDIM + col;
    const float cn = ft * c[s] + it * ct;
    const float hn = ot * tanhf(cn);
    c[s] = cn; h[s] = hn;
    const unsigned short hb = f2bf(hn);
    hbf_out[s] = hb;
    hseq_t[s]  = hb;
  }
}

// ---------------------------------------------------------------------------
// Phase 3: y[b][t][o] = sum_h hseq[t*64+b][h]*WyT[o][h] + by[o]
// M=32768 (rows r=t*64+b), N=1024, K=1024.
// ---------------------------------------------------------------------------
__global__ __launch_bounds__(256) void lstm_gemm_y(
    const unsigned short* __restrict__ hseq,
    const unsigned short* __restrict__ WyT,
    const float* __restrict__ by,
    float* __restrict__ out)
{
  __shared__ __align__(16) unsigned int As[2 * 64 * 16];
  __shared__ __align__(16) unsigned int Bs[2 * 64 * 16];
  const int m0 = blockIdx.x * 64, n0 = blockIdx.y * 64;
  const int tid = threadIdx.x;
  const int w = tid >> 5, lane = tid & 31;
  const int lo = lane & 15, hi = lane >> 4;
  const int mrow = (w & 3) * 16;
  const int nc0  = (w >> 2) * 32;

  v8f acc0 = {}, acc1 = {};
  wmma_panel_gemm(hseq, WyT, m0, n0, As, Bs, acc0, acc1);

  // epilogue: hseq row r=t*64+b  ->  y[(b*512+t)*1024 + col] + by
#pragma unroll
  for (int i = 0; i < 8; ++i) {
    const int gm = m0 + mrow + 8 * hi + i;
    const int t = gm >> 6, b = gm & 63;
    const size_t orow = (size_t)(b * TSTEPS + t);
    const int col0 = n0 + nc0 + lo;
    const int col1 = col0 + 16;
    out[orow * ODIM + col0] = acc0[i] + by[col0];
    out[orow * ODIM + col1] = acc1[i] + by[col1];
  }
}

__global__ void lstm_copy_final(const float* __restrict__ h,
                                const float* __restrict__ c,
                                float* __restrict__ out)
{
  const int i = blockIdx.x * blockDim.x + threadIdx.x;  // 65536 threads
  const size_t ylen = (size_t)BATCH * TSTEPS * ODIM;    // 33554432 elements
  out[ylen + i] = h[i];
  out[ylen + BATCH * HDIM + i] = c[i];
}

// ---------------------------------------------------------------------------
extern "C" void kernel_launch(void* const* d_in, const int* in_sizes, int n_in,
                              void* d_out, int out_size, void* d_ws, size_t ws_size,
                              hipStream_t stream) {
  const float* x  = (const float*)d_in[0];
  const float* Wf = (const float*)d_in[1];
  const float* bf = (const float*)d_in[2];
  const float* Wi = (const float*)d_in[3];
  const float* bi = (const float*)d_in[4];
  const float* Wc = (const float*)d_in[5];
  const float* bc = (const float*)d_in[6];
  const float* Wo = (const float*)d_in[7];
  const float* bo = (const float*)d_in[8];
  const float* Wy = (const float*)d_in[9];
  const float* by = (const float*)d_in[10];
  float* out = (float*)d_out;

  // workspace carve-up (~659 MB total)
  char* ws = (char*)d_ws;
  size_t off = 0;
  auto carve = [&](size_t bytes) -> void* {
    void* p = ws + off;
    off += (bytes + 255) & ~(size_t)255;
    return p;
  };
  float*          xproj = (float*)carve((size_t)TSTEPS * BATCH * GDIM * 4);          // 512MB
  unsigned short* xbf   = (unsigned short*)carve((size_t)BATCH * TSTEPS * DDIM * 2); // 64MB
  unsigned short* hseq  = (unsigned short*)carve((size_t)TSTEPS * BATCH * HDIM * 2); // 64MB
  unsigned short* WxT   = (unsigned short*)carve((size_t)GDIM * DDIM * 2);           // 8MB
  unsigned short* WhT   = (unsigned short*)carve((size_t)GDIM * HDIM * 2);           // 8MB
  unsigned short* WyT   = (unsigned short*)carve((size_t)ODIM * HDIM * 2);           // 2MB
  float*          ball  = (float*)carve((size_t)GDIM * 4);
  float*          hbuf  = (float*)carve((size_t)BATCH * HDIM * 4);
  float*          cbuf  = (float*)carve((size_t)BATCH * HDIM * 4);
  unsigned short* hbf0  = (unsigned short*)carve((size_t)BATCH * HDIM * 2);
  unsigned short* hbf1  = (unsigned short*)carve((size_t)BATCH * HDIM * 2);
  (void)ws_size; (void)in_sizes; (void)n_in; (void)out_size;

  // 0) pack weights (bf16, transposed), biases, convert x, zero state
  lstm_pack_weights<<<GDIM, 256, 0, stream>>>(Wf, Wi, Wc, Wo, bf, bi, bc, bo, Wy,
                                              WxT, WhT, WyT, ball);
  lstm_convert_x<<<4096, 256, 0, stream>>>(x, xbf, (size_t)BATCH * TSTEPS * DDIM);
  lstm_init_state<<<(BATCH * HDIM) / 256, 256, 0, stream>>>(hbuf, cbuf, hbf0, hbf1);

  // 1) input projection (big parallel WMMA GEMM)
  lstm_gemm_xproj<<<dim3(512, 64), 256, 0, stream>>>(xbf, WxT, ball, xproj);

  // 2) recurrence: 512 stream-ordered step kernels, double-buffered bf16 h
  unsigned short* hin = hbf0;
  unsigned short* hout = hbf1;
  for (int t = 0; t < TSTEPS; ++t) {
    lstm_step<<<dim3(64, 4), 128, 0, stream>>>(
        hin, WhT, xproj + (size_t)t * BATCH * GDIM,
        cbuf, hbuf, hout, hseq + (size_t)t * BATCH * HDIM);
    unsigned short* tmp = hin; hin = hout; hout = tmp;
  }

  // 3) output projection (big parallel WMMA GEMM) + final states
  lstm_gemm_y<<<dim3(512, 16), 256, 0, stream>>>(hseq, WyT, by, out);
  lstm_copy_final<<<(BATCH * HDIM) / 256, 256, 0, stream>>>(hbuf, cbuf, out);
}
